// Attention_80771154969277
// MI455X (gfx1250) — matrix operand order repaired and checked by reading, not compiled, not run on hardware
//
#include <hip/hip_runtime.h>
#include <hip/hip_bf16.h>
#include <stdint.h>

#define B_     4
#define N_     2048
#define DIM_   768
#define HEADS_ 12
#define HD_    64

typedef __attribute__((ext_vector_type(16))) _Float16 v16h;
typedef __attribute__((ext_vector_type(8)))  _Float16 v8h;
typedef __attribute__((ext_vector_type(4)))  _Float16 v4h;
typedef __attribute__((ext_vector_type(8)))  float    v8f;
typedef __attribute__((ext_vector_type(4)))  float    v4f;

// Async global->LDS copy of 16 bytes (CDNA5, tracked by ASYNCcnt).
// ldsoff = LDS byte offset (low 32 bits of the flat shared address).
__device__ __forceinline__ void async_copy_b128(unsigned ldsoff, const void* gaddr)
{
    asm volatile("global_load_async_to_lds_b128 %0, %1, off"
                 :: "v"(ldsoff), "v"(gaddr) : "memory");
}
__device__ __forceinline__ void async_wait0()
{
    asm volatile("s_wait_asynccnt 0x0" ::: "memory");
}
__device__ __forceinline__ unsigned lds_off(const void* p)
{
    return (unsigned)(uintptr_t)p;   // flat shared addr: low 32 bits = LDS offset
}

// ---------------------------------------------------------------------------
// Weight prep: dst[c*768 + k] = (f16) src[k*ld + c]   (transpose + convert)
// ---------------------------------------------------------------------------
__global__ void wconv_kernel(const float* __restrict__ src,
                             _Float16* __restrict__ dst, int ld)
{
    const int c = blockIdx.x;
    _Float16* drow = dst + (size_t)c * DIM_;
    for (int k = threadIdx.x; k < DIM_; k += blockDim.x)
        drow[k] = (_Float16)src[(size_t)k * ld + c];
}

// x (f32) -> Xh (f16), vectorized; read once instead of 9x as f32.
__global__ void xconv_kernel(const float* __restrict__ x,
                             _Float16* __restrict__ Xh, int n4)
{
    for (int i = blockIdx.x * blockDim.x + threadIdx.x; i < n4;
         i += gridDim.x * blockDim.x) {
        const v4f f = *(const v4f*)&x[(size_t)i * 4];
        v4h hv; hv[0] = (_Float16)f[0]; hv[1] = (_Float16)f[1];
        hv[2] = (_Float16)f[2]; hv[3] = (_Float16)f[3];
        *(v4h*)&Xh[(size_t)i * 4] = hv;
    }
}

// ---------------------------------------------------------------------------
// Kernel 1: fused QKV projection.  Y = Xh @ [Wq | Wkv]  (8192 x 2304), f16 WMMA.
// Block = 128 threads (4 waves), tile 64 rows x 256 cols, K stepped by 32.
// A tile staged via async global->LDS DMA; B frags are contiguous 32B v16h
// loads from pre-transposed f16 weights (L2-hot).
// ---------------------------------------------------------------------------
__global__ void __launch_bounds__(128, 1)
qkv_gemm_kernel(const _Float16* __restrict__ Xh,
                const _Float16* __restrict__ WallT,   // [2304][768] f16
                float* __restrict__ Qr,
                float* __restrict__ Kr,
                _Float16* __restrict__ Vh)
{
    __shared__ _Float16 Alds[64][48];   // stride 96B -> 16B-aligned frags

    const int tid   = threadIdx.x;
    const int wave  = tid >> 5;
    const int lane  = tid & 31;
    const int laneM = lane & 15;
    const int laneH = lane >> 4;
    const int rowbase = blockIdx.y * 64;
    const int colbase = blockIdx.x * 256 + wave * 64;

    v8f acc[4][4];
    #pragma unroll
    for (int rt = 0; rt < 4; ++rt)
        #pragma unroll
        for (int ct = 0; ct < 4; ++ct) { v8f z = {}; acc[rt][ct] = z; }

    for (int k0 = 0; k0 < DIM_; k0 += 32) {
        // async stage A tile (64 rows x 32 k, f16): 2 x b128 per thread
        #pragma unroll
        for (int j = 0; j < 2; ++j) {
            const int idx = tid + 128 * j;          // 0..255
            const int r = idx >> 2, cq = (idx & 3) * 8;
            async_copy_b128(lds_off(&Alds[r][cq]),
                            &Xh[(size_t)(rowbase + r) * DIM_ + k0 + cq]);
        }
        async_wait0();
        __syncthreads();

        v16h afrag[4];
        #pragma unroll
        for (int rt = 0; rt < 4; ++rt) {
            const int m  = rt * 16 + laneM;
            const int kb = laneH * 16;
            const v8h lo = *(const v8h*)&Alds[m][kb];
            const v8h hi = *(const v8h*)&Alds[m][kb + 8];
            #pragma unroll
            for (int i = 0; i < 8; ++i) { afrag[rt][i] = lo[i]; afrag[rt][8 + i] = hi[i]; }
        }

        #pragma unroll
        for (int ct = 0; ct < 4; ++ct) {
            const int gcol = colbase + ct * 16 + laneM;
            const v16h bfrag =
                *(const v16h*)&WallT[(size_t)gcol * DIM_ + k0 + laneH * 16];
            #pragma unroll
            for (int rt = 0; rt < 4; ++rt)
                acc[rt][ct] = __builtin_amdgcn_wmma_f32_16x16x32_f16(
                    false, afrag[rt], false, bfrag, (short)0, acc[rt][ct],
                    false, false);
        }
        __syncthreads();
    }

    // scatter to Q/K (f32, [B,H,N,HD]) and V (f16); lanes consecutive in d.
    #pragma unroll
    for (int ct = 0; ct < 4; ++ct) {
        const int gcol = colbase + ct * 16 + laneM;
        #pragma unroll
        for (int rt = 0; rt < 4; ++rt) {
            #pragma unroll
            for (int r = 0; r < 8; ++r) {
                const int grow = rowbase + rt * 16 + laneH * 8 + r;
                const int b = grow >> 11;
                const int n = grow & 2047;
                const float val = acc[rt][ct][r];
                if (gcol < DIM_) {
                    const int head = gcol >> 6, d = gcol & 63;
                    Qr[(((size_t)b * HEADS_ + head) * N_ + n) * HD_ + d] = val;
                } else if (gcol < 2 * DIM_) {
                    const int cc = gcol - DIM_;
                    const int head = cc >> 6, d = cc & 63;
                    Kr[(((size_t)b * HEADS_ + head) * N_ + n) * HD_ + d] = val;
                } else {
                    const int cc = gcol - 2 * DIM_;
                    const int head = cc >> 6, d = cc & 63;
                    Vh[(((size_t)b * HEADS_ + head) * N_ + n) * HD_ + d] = (_Float16)val;
                }
            }
        }
    }
}

// ---------------------------------------------------------------------------
// Kernel 2: per-head RMS-norm + 2D rotary, fp32 -> f16.
// ---------------------------------------------------------------------------
__global__ void normrope_kernel(const float* __restrict__ Qr,
                                const float* __restrict__ Kr,
                                const int*   __restrict__ positions,
                                const float* __restrict__ qn_w,
                                const float* __restrict__ kn_w,
                                const float* __restrict__ inv_freq,
                                _Float16* __restrict__ Qh,
                                _Float16* __restrict__ Kh)
{
    __shared__ float sh[64];
    __shared__ float row[64];

    const int d   = threadIdx.x;
    const int idx = blockIdx.x;            // (b*N + n)*HEADS + h
    const int h   = idx % HEADS_;
    const int bn  = idx / HEADS_;          // b*N + n
    const bool isK = (blockIdx.y != 0);

    const size_t base = ((size_t)(bn / N_) * HEADS_ + h) * N_ + (bn % N_);
    const float* src = (isK ? Kr : Qr) + base * HD_;
    const float* w   = isK ? kn_w : qn_w;
    _Float16*   dst  = (isK ? Kh : Qh) + base * HD_;

    const float v = src[d];
    sh[d] = v * v;
    __syncthreads();
    for (int s = 32; s > 0; s >>= 1) {
        if (d < s) sh[d] += sh[d + s];
        __syncthreads();
    }
    const float rs = rsqrtf(sh[0] * (1.0f / 64.0f) + 1e-6f);
    row[d] = v * rs * w[d];
    __syncthreads();

    const int halfsel = d >> 5;
    const int local   = d & 31;
    const int i       = local >> 1;
    const float pos   = (float)positions[(size_t)bn * 2 + halfsel];
    const float ang   = pos * inv_freq[i];
    const float c = cosf(ang), sn = sinf(ang);
    const int   pbase = halfsel * 32;
    const float xe = row[pbase + 2 * i];
    const float xo = row[pbase + 2 * i + 1];
    const float outv = ((local & 1) == 0) ? (xe * c - xo * sn)
                                          : (xe * sn + xo * c);
    dst[d] = (_Float16)outv;
}

// ---------------------------------------------------------------------------
// Kernel 3: flash attention per (b,h).  64 queries/block (4 waves x 16),
// 32-key chunks.  K staged via async DMA (row-major, aligned v16h frags),
// V transposed in LDS (aligned v16h frags), P restaged via aligned v8h.
// ---------------------------------------------------------------------------
__global__ void __launch_bounds__(128, 1)
flash_attn_kernel(const _Float16* __restrict__ Qh,
                  const _Float16* __restrict__ Kh,
                  const _Float16* __restrict__ Vh,
                  _Float16* __restrict__ Oh)
{
    __shared__ _Float16 Klds[32][80];       // row stride 160B (32B-aligned frags)
    __shared__ _Float16 VldsT[64][48];      // [d][key], stride 96B
    __shared__ _Float16 Plds[4][16][48];    // per-wave P tile

    const int tid   = threadIdx.x;
    const int wave  = tid >> 5;
    const int lane  = tid & 31;
    const int laneM = lane & 15;
    const int laneH = lane >> 4;
    const int bh    = blockIdx.y;
    const int b     = bh / HEADS_;
    const int h     = bh % HEADS_;
    const int myq   = blockIdx.x * 64 + wave * 16;

    const _Float16* Qp = Qh + (size_t)bh * N_ * HD_;
    const _Float16* Kp = Kh + (size_t)bh * N_ * HD_;
    const _Float16* Vp = Vh + (size_t)bh * N_ * HD_;

    // Q A-fragments (held for whole key loop): 4 aligned v8h global loads
    v16h qa0, qa1;
    {
        const int m  = myq + laneM;
        const int kb = laneH * 16;
        const _Float16* qrow = Qp + (size_t)m * HD_;
        const v8h a0 = *(const v8h*)&qrow[kb];
        const v8h a1 = *(const v8h*)&qrow[kb + 8];
        const v8h a2 = *(const v8h*)&qrow[32 + kb];
        const v8h a3 = *(const v8h*)&qrow[32 + kb + 8];
        #pragma unroll
        for (int i = 0; i < 8; ++i) {
            qa0[i] = a0[i]; qa0[8 + i] = a1[i];
            qa1[i] = a2[i]; qa1[8 + i] = a3[i];
        }
    }

    v8f o[4];
    #pragma unroll
    for (int dt = 0; dt < 4; ++dt) { v8f z = {}; o[dt] = z; }
    float mrow[8], lrow[8];
    #pragma unroll
    for (int r = 0; r < 8; ++r) { mrow[r] = -1e30f; lrow[r] = 0.0f; }
    const float scale = 0.125f;

    // per-thread chunk-load coordinates (16B granules)
    const int ckk = tid >> 3;              // 0..15 (+16 with j)
    const int cdq = (tid & 7) * 8;

    for (int kc = 0; kc < N_; kc += 32) {
        // K: async DMA to LDS; V: load + transpose store
        #pragma unroll
        for (int j = 0; j < 2; ++j) {
            const int kk = ckk + 16 * j;
            async_copy_b128(lds_off(&Klds[kk][cdq]),
                            &Kp[(size_t)(kc + kk) * HD_ + cdq]);
            const v8h vv = *(const v8h*)&Vp[(size_t)(kc + kk) * HD_ + cdq];
            #pragma unroll
            for (int i = 0; i < 8; ++i) VldsT[cdq + i][kk] = vv[i];
        }
        // prefetch next chunk while this one is consumed
        if (kc + 32 < N_) {
            __builtin_prefetch(&Kp[(size_t)(kc + 32 + ckk) * HD_ + cdq], 0, 1);
            __builtin_prefetch(&Vp[(size_t)(kc + 32 + ckk) * HD_ + cdq], 0, 1);
        }
        async_wait0();
        __syncthreads();

        // S = Q K^T : two 16-key tiles, 2 WMMA each over HD=64
        v8f s[2];
        #pragma unroll
        for (int t = 0; t < 2; ++t) {
            const int key = t * 16 + laneM;
            const int kb  = laneH * 16;
            const v16h b0 = *(const v16h*)&Klds[key][kb];
            const v16h b1 = *(const v16h*)&Klds[key][32 + kb];
            v8f sz = {};
            sz = __builtin_amdgcn_wmma_f32_16x16x32_f16(false, qa0, false, b0,
                                                        (short)0, sz, false, false);
            sz = __builtin_amdgcn_wmma_f32_16x16x32_f16(false, qa1, false, b1,
                                                        (short)0, sz, false, false);
            s[t] = sz;
        }

        // online softmax update (C-layout: row = r + 8*laneH, col = laneM)
        float alpha[8];
        #pragma unroll
        for (int r = 0; r < 8; ++r) {
            const float s0 = s[0][r] * scale;
            const float s1 = s[1][r] * scale;
            float mx = fmaxf(s0, s1);
            #pragma unroll
            for (int msk = 1; msk < 16; msk <<= 1)
                mx = fmaxf(mx, __shfl_xor(mx, msk, 32));
            const float mnew = fmaxf(mrow[r], mx);
            alpha[r] = __expf(mrow[r] - mnew);
            mrow[r]  = mnew;
            const float p0 = __expf(s0 - mnew);
            const float p1 = __expf(s1 - mnew);
            s[0][r] = p0;
            s[1][r] = p1;
            float ps = p0 + p1;
            #pragma unroll
            for (int msk = 1; msk < 16; msk <<= 1)
                ps += __shfl_xor(ps, msk, 32);
            lrow[r] = lrow[r] * alpha[r] + ps;
        }
        #pragma unroll
        for (int dt = 0; dt < 4; ++dt)
            #pragma unroll
            for (int r = 0; r < 8; ++r)
                o[dt][r] *= alpha[r];

        // stage P (C-layout -> LDS), reload as A fragment (aligned v8h)
        #pragma unroll
        for (int t = 0; t < 2; ++t)
            #pragma unroll
            for (int r = 0; r < 8; ++r)
                Plds[wave][laneH * 8 + r][t * 16 + laneM] = (_Float16)s[t][r];

        v16h pa;
        {
            const int kb = laneH * 16;
            const v8h lo = *(const v8h*)&Plds[wave][laneM][kb];
            const v8h hi = *(const v8h*)&Plds[wave][laneM][kb + 8];
            #pragma unroll
            for (int i = 0; i < 8; ++i) { pa[i] = lo[i]; pa[8 + i] = hi[i]; }
        }

        // O += P @ V : bv fragment = one aligned v16h load from VldsT
        #pragma unroll
        for (int dt = 0; dt < 4; ++dt) {
            const v16h bv = *(const v16h*)&VldsT[dt * 16 + laneM][laneH * 16];
            o[dt] = __builtin_amdgcn_wmma_f32_16x16x32_f16(false, pa, false, bv,
                                                           (short)0, o[dt],
                                                           false, false);
        }
        __syncthreads();
    }

    #pragma unroll
    for (int r = 0; r < 8; ++r) lrow[r] = 1.0f / lrow[r];
    #pragma unroll
    for (int dt = 0; dt < 4; ++dt) {
        #pragma unroll
        for (int r = 0; r < 8; ++r) {
            const int n   = myq + laneH * 8 + r;
            const int col = h * HD_ + dt * 16 + laneM;
            Oh[((size_t)b * N_ + n) * DIM_ + col] = (_Float16)(o[dt][r] * lrow[r]);
        }
    }
}

// ---------------------------------------------------------------------------
// Kernel 4: out = O @ Wp + bp   (A f16 async-staged, B pre-transposed f16)
// ---------------------------------------------------------------------------
__global__ void __launch_bounds__(128, 1)
out_gemm_kernel(const _Float16* __restrict__ Oh,
                const _Float16* __restrict__ WpT,   // [768][768] f16
                const float* __restrict__ bp,
                float* __restrict__ out)
{
    __shared__ _Float16 Alds[64][48];

    const int tid   = threadIdx.x;
    const int wave  = tid >> 5;
    const int lane  = tid & 31;
    const int laneM = lane & 15;
    const int laneH = lane >> 4;
    const int rowbase = blockIdx.y * 64;
    const int colbase = blockIdx.x * 256 + wave * 64;

    v8f acc[4][4];
    #pragma unroll
    for (int rt = 0; rt < 4; ++rt)
        #pragma unroll
        for (int ct = 0; ct < 4; ++ct) { v8f z = {}; acc[rt][ct] = z; }

    for (int k0 = 0; k0 < DIM_; k0 += 32) {
        #pragma unroll
        for (int j = 0; j < 2; ++j) {
            const int idx = tid + 128 * j;          // 0..255
            const int r = idx >> 2, cq = (idx & 3) * 8;
            async_copy_b128(lds_off(&Alds[r][cq]),
                            &Oh[(size_t)(rowbase + r) * DIM_ + k0 + cq]);
        }
        async_wait0();
        __syncthreads();

        v16h afrag[4];
        #pragma unroll
        for (int rt = 0; rt < 4; ++rt) {
            const int m  = rt * 16 + laneM;
            const int kb = laneH * 16;
            const v8h lo = *(const v8h*)&Alds[m][kb];
            const v8h hi = *(const v8h*)&Alds[m][kb + 8];
            #pragma unroll
            for (int i = 0; i < 8; ++i) { afrag[rt][i] = lo[i]; afrag[rt][8 + i] = hi[i]; }
        }

        #pragma unroll
        for (int ct = 0; ct < 4; ++ct) {
            const int gcol = colbase + ct * 16 + laneM;
            const v16h bfrag =
                *(const v16h*)&WpT[(size_t)gcol * DIM_ + k0 + laneH * 16];
            #pragma unroll
            for (int rt = 0; rt < 4; ++rt)
                acc[rt][ct] = __builtin_amdgcn_wmma_f32_16x16x32_f16(
                    false, afrag[rt], false, bfrag, (short)0, acc[rt][ct],
                    false, false);
        }
        __syncthreads();
    }

    #pragma unroll
    for (int ct = 0; ct < 4; ++ct) {
        const int gcol = colbase + ct * 16 + laneM;
        const float bias = bp[gcol];
        #pragma unroll
        for (int rt = 0; rt < 4; ++rt) {
            #pragma unroll
            for (int r = 0; r < 8; ++r) {
                const int grow = rowbase + rt * 16 + laneH * 8 + r;
                out[(size_t)grow * DIM_ + gcol] = acc[rt][ct][r] + bias;
            }
        }
    }
}

// ---------------------------------------------------------------------------
extern "C" void kernel_launch(void* const* d_in, const int* in_sizes, int n_in,
                              void* d_out, int out_size, void* d_ws, size_t ws_size,
                              hipStream_t stream)
{
    (void)in_sizes; (void)n_in; (void)out_size; (void)ws_size;

    const float* x         = (const float*)d_in[0];
    const int*   positions = (const int*)  d_in[1];
    const float* Wq        = (const float*)d_in[2];
    const float* Wkv       = (const float*)d_in[3];
    const float* Wp        = (const float*)d_in[4];
    const float* bp        = (const float*)d_in[5];
    const float* qn_w      = (const float*)d_in[6];
    const float* kn_w      = (const float*)d_in[7];
    const float* inv_freq  = (const float*)d_in[8];
    float*       out       = (float*)d_out;

    const size_t ELEMS = (size_t)B_ * N_ * DIM_;   // 6,291,456
    char* w = (char*)d_ws;
    float*    Qr    = (float*)w;     w += ELEMS * sizeof(float);
    float*    Kr    = (float*)w;     w += ELEMS * sizeof(float);
    _Float16* Vh    = (_Float16*)w;  w += ELEMS * sizeof(_Float16);
    _Float16* Qh    = (_Float16*)w;  w += ELEMS * sizeof(_Float16);
    _Float16* Kh    = (_Float16*)w;  w += ELEMS * sizeof(_Float16);
    _Float16* Oh    = (_Float16*)w;  w += ELEMS * sizeof(_Float16);
    _Float16* Xh    = (_Float16*)w;  w += ELEMS * sizeof(_Float16);
    _Float16* WallT = (_Float16*)w;  w += (size_t)3 * DIM_ * DIM_ * sizeof(_Float16);
    _Float16* WpT   = (_Float16*)w;  w += (size_t)DIM_ * DIM_ * sizeof(_Float16);

    // 0) one-shot conversions (weights transposed; x to f16)
    wconv_kernel<<<dim3(DIM_),     256, 0, stream>>>(Wq,  WallT,                       DIM_);
    wconv_kernel<<<dim3(2 * DIM_), 256, 0, stream>>>(Wkv, WallT + (size_t)DIM_ * DIM_, 2 * DIM_);
    wconv_kernel<<<dim3(DIM_),     256, 0, stream>>>(Wp,  WpT,                         DIM_);
    xconv_kernel<<<dim3(2048),     256, 0, stream>>>(x, Xh, (int)(ELEMS / 4));

    // 1) QKV projection (WMMA, async LDS staging)
    qkv_gemm_kernel<<<dim3(9, (B_ * N_) / 64), 128, 0, stream>>>(
        Xh, WallT, Qr, Kr, Vh);

    // 2) RMS-norm + rotary on q,k
    normrope_kernel<<<dim3(B_ * N_ * HEADS_, 2), 64, 0, stream>>>(
        Qr, Kr, positions, qn_w, kn_w, inv_freq, Qh, Kh);

    // 3) flash attention (WMMA, async K staging + prefetch)
    flash_attn_kernel<<<dim3(N_ / 64, B_ * HEADS_), 128, 0, stream>>>(
        Qh, Kh, Vh, Oh);

    // 4) output projection + bias (WMMA, async LDS staging)
    out_gemm_kernel<<<dim3(3, (B_ * N_) / 64), 128, 0, stream>>>(
        Oh, WpT, bp, out);
}
